// MultiHeadAttention_15058155339996
// MI455X (gfx1250) — compile-verified, matrix operand order checked by reading, not run on hardware
//
#include <hip/hip_runtime.h>

typedef __bf16 bf16;
typedef __attribute__((ext_vector_type(16))) __bf16 v16bf;
typedef __attribute__((ext_vector_type(8)))  float  v8f;
typedef __attribute__((ext_vector_type(4)))  unsigned int v4u;

#define B_  4
#define N_  2048
#define C_  768
#define H_  12
#define D_  64
#define C3_ 2304

union FragU { v4u u[2]; v16bf v; };

// Load a 16x32 bf16 fragment (A-layout per CDNA5 ISA 7.12.2):
// lanes 0-15: row = lane,    k = {0..7, 16..23}
// lanes 16-31: row = lane-16, k = {8..15, 24..31}
// -> two contiguous 16-byte loads per lane.
// The same mapping serves B-fragments (row index = N column of B), since
// W / K^T / V^T tiles are stored with the WMMA K dim contiguous per row.
__device__ __forceinline__ v16bf load_frag(const bf16* __restrict__ base, int ld) {
  const int lane = threadIdx.x & 31;
  const int r  = lane & 15;
  const int kh = (lane >> 4) << 3;
  const bf16* p = base + r * ld + kh;
  FragU f;
  f.u[0] = *reinterpret_cast<const v4u*>(p);
  f.u[1] = *reinterpret_cast<const v4u*>(p + 16);
  return f.v;
}

__device__ __forceinline__ v8f wmma_bf16(v16bf a, v16bf b, v8f c) {
  return __builtin_amdgcn_wmma_f32_16x16x32_bf16(false, a, false, b, (short)0, c, false, false);
}

__global__ void cvt_f32_bf16(const float* __restrict__ in, bf16* __restrict__ out, int n) {
  int i = blockIdx.x * blockDim.x + threadIdx.x;
  if (i < n) out[i] = (bf16)in[i];
}

// C[M=8192, N=2304] = Xbf[M,768] @ Wbf[N,768]^T + bias, scattered to
// Q[b,h,n,d] (scaled by 1/8), K[b,h,n,d], Vt[b,h,d,n]  (all bf16)
// Each wave: 32(M) x 64(N) tile -> 8 accumulators, A reused across N.
__global__ void qkv_gemm(const bf16* __restrict__ xb, const bf16* __restrict__ wb,
                         const float* __restrict__ bias,
                         bf16* __restrict__ Qb, bf16* __restrict__ Kb, bf16* __restrict__ Vt) {
  const int lane = threadIdx.x & 31;
  const int wave = threadIdx.x >> 5;
  const int gw   = blockIdx.x * 4 + wave;
  const int NT   = C3_ / 64;                 // 36 column tiles of 64
  const int mt   = gw / NT, nt = gw % NT;
  const int row0 = mt * 32, col0 = nt * 64;

  v8f acc[2][4] = {};
  for (int k = 0; k < C_; k += 32) {
    const v16bf a0 = load_frag(xb + (size_t)row0 * C_ + k,        C_);
    const v16bf a1 = load_frag(xb + (size_t)(row0 + 16) * C_ + k, C_);
#pragma unroll
    for (int j = 0; j < 4; ++j) {
      const v16bf b = load_frag(wb + (size_t)(col0 + j * 16) * C_ + k, C_);
      acc[0][j] = wmma_bf16(a0, b, acc[0][j]);
      acc[1][j] = wmma_bf16(a1, b, acc[1][j]);
    }
  }

  const int n  = lane & 15;
  const int m0 = (lane >> 4) << 3;
#pragma unroll
  for (int j = 0; j < 4; ++j) {
    const int c3 = col0 + j * 16 + n;
    const int t  = c3 / C_;
    const int r  = c3 - t * C_;
    const int h  = r >> 6;
    const int d  = r & 63;
    const float bv = bias[c3];
#pragma unroll
    for (int mi = 0; mi < 2; ++mi) {
#pragma unroll
      for (int i = 0; i < 8; ++i) {
        const int gm = row0 + mi * 16 + m0 + i;
        const int b_ = gm >> 11;                 // / N_
        const int nn = gm & (N_ - 1);
        const float v = acc[mi][j][i] + bv;
        const size_t hb = (size_t)(b_ * H_ + h);
        if (t == 0)       Qb[(hb * N_ + nn) * D_ + d] = (bf16)(v * 0.125f);  // fold 1/sqrt(64)
        else if (t == 1)  Kb[(hb * N_ + nn) * D_ + d] = (bf16)v;
        else              Vt[(hb * D_ + d) * N_ + nn] = (bf16)v;             // V transposed
      }
    }
  }
}

// Flash attention: one wave = 16 query rows of one (b,h); loop 32-key tiles.
__global__ void attn_flash(const bf16* __restrict__ Qb, const bf16* __restrict__ Kb,
                           const bf16* __restrict__ Vt, bf16* __restrict__ Ob) {
  __shared__ __align__(16) bf16 plds[4][16 * 32];   // private P tile per wave
  const int lane = threadIdx.x & 31;
  const int wave = threadIdx.x >> 5;
  const int gw   = blockIdx.x * 4 + wave;
  const int bh   = gw >> 7;                 // / (N_/16)
  const int q0   = (gw & 127) * 16;

  const bf16* Qh = Qb + (size_t)bh * N_ * D_;
  const bf16* Kh = Kb + (size_t)bh * N_ * D_;
  const bf16* Vh = Vt + (size_t)bh * D_ * N_;

  const v16bf qa0 = load_frag(Qh + (size_t)q0 * D_,      D_);
  const v16bf qa1 = load_frag(Qh + (size_t)q0 * D_ + 32, D_);

  float rm[8], rl[8];
#pragma unroll
  for (int i = 0; i < 8; ++i) { rm[i] = -3.0e38f; rl[i] = 0.f; }
  v8f o[4] = {};

  const int n  = lane & 15;
  const int m0 = (lane >> 4) << 3;
  const int kh = (lane >> 4) << 3;
  bf16* myp = &plds[wave][0];

  for (int kt = 0; kt < N_; kt += 32) {
    v8f s0 = {}, s1 = {};
    v16bf kb;
    kb = load_frag(Kh + (size_t)kt * D_,             D_); s0 = wmma_bf16(qa0, kb, s0);
    kb = load_frag(Kh + (size_t)kt * D_ + 32,        D_); s0 = wmma_bf16(qa1, kb, s0);
    kb = load_frag(Kh + (size_t)(kt + 16) * D_,      D_); s1 = wmma_bf16(qa0, kb, s1);
    kb = load_frag(Kh + (size_t)(kt + 16) * D_ + 32, D_); s1 = wmma_bf16(qa1, kb, s1);

    // online softmax; row statistics live per 16-lane half (C-fragment striping)
#pragma unroll
    for (int i = 0; i < 8; ++i) {
      float tm = fmaxf(s0[i], s1[i]);
      tm = fmaxf(tm, __shfl_xor(tm, 1));
      tm = fmaxf(tm, __shfl_xor(tm, 2));
      tm = fmaxf(tm, __shfl_xor(tm, 4));
      tm = fmaxf(tm, __shfl_xor(tm, 8));
      const float nm = fmaxf(rm[i], tm);
      const float p0 = __expf(s0[i] - nm);
      const float p1 = __expf(s1[i] - nm);
      float rs = p0 + p1;
      rs += __shfl_xor(rs, 1);
      rs += __shfl_xor(rs, 2);
      rs += __shfl_xor(rs, 4);
      rs += __shfl_xor(rs, 8);
      const float alpha = __expf(rm[i] - nm);
      rl[i] = rl[i] * alpha + rs;
      rm[i] = nm;
#pragma unroll
      for (int j = 0; j < 4; ++j) o[j][i] *= alpha;
      const int m = m0 + i;
      myp[m * 32 + n]      = (bf16)p0;   // C-layout -> LDS (row-major 16x32)
      myp[m * 32 + 16 + n] = (bf16)p1;
    }

    // re-read P in A-fragment layout (DS ops are in-order within a wave)
    FragU pf;
    pf.u[0] = *reinterpret_cast<const v4u*>(&myp[(lane & 15) * 32 + kh]);
    pf.u[1] = *reinterpret_cast<const v4u*>(&myp[(lane & 15) * 32 + kh + 16]);

#pragma unroll
    for (int j = 0; j < 4; ++j) {
      v16bf vb = load_frag(Vh + (size_t)(j * 16) * N_ + kt, N_);  // V^T: contiguous in key dim
      o[j] = wmma_bf16(pf.v, vb, o[j]);
    }
  }

  const int b_ = bh / H_;
  const int h  = bh % H_;
#pragma unroll
  for (int j = 0; j < 4; ++j) {
#pragma unroll
    for (int i = 0; i < 8; ++i) {
      const int q = q0 + m0 + i;
      const float v = o[j][i] / rl[i];
      Ob[((size_t)(b_ * N_ + q)) * C_ + h * D_ + j * 16 + n] = (bf16)v;
    }
  }
}

// out[M=8192, 768] (f32) = Ab[M,768] @ Wb[768,768]^T + bias
// Each wave: 32(M) x 64(N) tile.
__global__ void proj_gemm(const bf16* __restrict__ ab, const bf16* __restrict__ wb,
                          const float* __restrict__ bias, float* __restrict__ out) {
  const int lane = threadIdx.x & 31;
  const int wave = threadIdx.x >> 5;
  const int gw   = blockIdx.x * 4 + wave;
  const int NT   = C_ / 64;                  // 12
  const int mt   = gw / NT, nt = gw % NT;
  const int row0 = mt * 32, col0 = nt * 64;

  v8f acc[2][4] = {};
  for (int k = 0; k < C_; k += 32) {
    const v16bf a0 = load_frag(ab + (size_t)row0 * C_ + k,        C_);
    const v16bf a1 = load_frag(ab + (size_t)(row0 + 16) * C_ + k, C_);
#pragma unroll
    for (int j = 0; j < 4; ++j) {
      const v16bf b = load_frag(wb + (size_t)(col0 + j * 16) * C_ + k, C_);
      acc[0][j] = wmma_bf16(a0, b, acc[0][j]);
      acc[1][j] = wmma_bf16(a1, b, acc[1][j]);
    }
  }

  const int n  = lane & 15;
  const int m0 = (lane >> 4) << 3;
#pragma unroll
  for (int j = 0; j < 4; ++j) {
    const int c = col0 + j * 16 + n;
    const float bv = bias[c];
#pragma unroll
    for (int mi = 0; mi < 2; ++mi) {
#pragma unroll
      for (int i = 0; i < 8; ++i) {
        const int gm = row0 + mi * 16 + m0 + i;
        out[(size_t)gm * C_ + c] = acc[mi][j][i] + bv;
      }
    }
  }
}

static inline size_t align256(size_t x) { return (x + 255) & ~(size_t)255; }

extern "C" void kernel_launch(void* const* d_in, const int* in_sizes, int n_in,
                              void* d_out, int out_size, void* d_ws, size_t ws_size,
                              hipStream_t stream) {
  const float* x      = (const float*)d_in[0];
  const float* qkv_w  = (const float*)d_in[1];
  const float* qkv_b  = (const float*)d_in[2];
  const float* proj_w = (const float*)d_in[3];
  const float* proj_b = (const float*)d_in[4];
  float* out = (float*)d_out;

  const size_t nX  = (size_t)B_ * N_ * C_;   // 6,291,456 (== B*H*N*D)
  const size_t nWq = (size_t)C3_ * C_;       // 1,769,472
  const size_t nWp = (size_t)C_ * C_;        //   589,824

  char* ws = (char*)d_ws;
  bf16* xb  = (bf16*)ws; ws += align256(nX  * 2);
  bf16* wqb = (bf16*)ws; ws += align256(nWq * 2);
  bf16* wpb = (bf16*)ws; ws += align256(nWp * 2);
  bf16* Qb  = (bf16*)ws; ws += align256(nX  * 2);
  bf16* Kb  = (bf16*)ws; ws += align256(nX  * 2);
  bf16* Vt  = (bf16*)ws; ws += align256(nX  * 2);
  bf16* Ob  = (bf16*)ws; ws += align256(nX  * 2);

  cvt_f32_bf16<<<(unsigned)((nX  + 255) / 256), 256, 0, stream>>>(x,      xb,  (int)nX);
  cvt_f32_bf16<<<(unsigned)((nWq + 255) / 256), 256, 0, stream>>>(qkv_w,  wqb, (int)nWq);
  cvt_f32_bf16<<<(unsigned)((nWp + 255) / 256), 256, 0, stream>>>(proj_w, wpb, (int)nWp);

  // 256 M-tiles(32) * 36 N-tiles(64) = 9216 waves / 4 per block
  qkv_gemm<<<2304, 128, 0, stream>>>(xb, wqb, qkv_b, Qb, Kb, Vt);
  // 48 (b,h) * 128 query tiles = 6144 waves / 4 per block
  attn_flash<<<1536, 128, 0, stream>>>(Qb, Kb, Vt, Ob);
  // 256 M-tiles(32) * 12 N-tiles(64) = 3072 waves / 4 per block
  proj_gemm<<<768, 128, 0, stream>>>(Ob, wpb, proj_b, out);
}
